// MultiHeadAttention_54563264528920
// MI455X (gfx1250) — matrix operand order repaired and checked by reading, not compiled, not run on hardware
//
#include <hip/hip_runtime.h>

// ---------------------------------------------------------------------------
// MHA forward for gfx1250 (MI455X, wave32, WMMA + async-to-LDS).
// B=2 S=2048 H=1024 NH=16 HD=64. All matmuls in bf16 WMMA w/ f32 accumulate.
// ---------------------------------------------------------------------------

typedef __attribute__((ext_vector_type(16))) __bf16 v16bf;
typedef __attribute__((ext_vector_type(8)))  float  v8f;
typedef __attribute__((ext_vector_type(4)))  int    v4i;

#define B_   2
#define S_   2048
#define H_   1024
#define NH_  16
#define HD_  64
#define M_   (B_ * S_)   // 4096 rows for all GEMMs

// CDNA5 async global->LDS path (ASYNCcnt-tracked), if the toolchain has it.
#define GEMM_ASYNC (__has_builtin(__builtin_amdgcn_global_load_async_to_lds_b128) && \
                    __has_builtin(__builtin_amdgcn_s_wait_asynccnt))
#define AS3 __attribute__((address_space(3)))

// Load a 16-element bf16 fragment as two 16-byte chunks (global or LDS).
__device__ __forceinline__ v16bf ld_frag2(const __bf16* p0, const __bf16* p1) {
  union { uint4 q[2]; v16bf v; } u;
  u.q[0] = *(const uint4*)p0;
  u.q[1] = *(const uint4*)p1;
  return u.v;
}

__device__ __forceinline__ float rmax16(float v) {
  v = fmaxf(v, __shfl_xor(v, 8, 32));
  v = fmaxf(v, __shfl_xor(v, 4, 32));
  v = fmaxf(v, __shfl_xor(v, 2, 32));
  v = fmaxf(v, __shfl_xor(v, 1, 32));
  return v;
}
__device__ __forceinline__ float rsum16(float v) {
  v += __shfl_xor(v, 8, 32);
  v += __shfl_xor(v, 4, 32);
  v += __shfl_xor(v, 2, 32);
  v += __shfl_xor(v, 1, 32);
  return v;
}

// ---------------------------------------------------------------------------
// f32 -> bf16 conversion, 4 elements / thread
// ---------------------------------------------------------------------------
__global__ void cvt_f32_bf16_x4(const float* __restrict__ in,
                                __bf16* __restrict__ out, int n4) {
  int i = blockIdx.x * blockDim.x + threadIdx.x;
  if (i >= n4) return;
  float4 f = ((const float4*)in)[i];
  union { __bf16 h[4]; ushort4 u; } cv;
  cv.h[0] = (__bf16)f.x; cv.h[1] = (__bf16)f.y;
  cv.h[2] = (__bf16)f.z; cv.h[3] = (__bf16)f.w;
  ((ushort4*)out)[i] = cv.u;
}

// ---------------------------------------------------------------------------
// GEMM: C[m,n] = sum_k A[m,k] * W[n,k] + bias[n]   (torch Linear: x @ W.T + b)
// A:[M,K] bf16 row-major, W:[N,K] bf16 row-major (both contract over contiguous K).
// Block tile 128x128, BK=64, 8 waves; wave computes 32x64 (2x4 WMMA tiles).
// Double-buffered LDS; staging via GLOBAL_LOAD_ASYNC_TO_LDS_B128 when available
// (next tile's loads overlap this tile's 16 WMMAs), else plain load/store.
// MODE 0: bf16 out [M,N] (Q,K). MODE 1: bf16 out transposed per head
//         [B,NH,HD,S] (V -> Vt, makes P@V fragments contiguous).
// MODE 2: f32 out [M,N] (final projection -> d_out).
// ---------------------------------------------------------------------------
template <int MODE>
__global__ __launch_bounds__(256) void gemm_xwT(
    const __bf16* __restrict__ A, const __bf16* __restrict__ W,
    const float* __restrict__ bias, void* __restrict__ OutP,
    int M, int N, int K) {
  constexpr int BM = 128, BN = 128, BK = 64, LDK = 72;  // 72 halves: 16B rows, bank-spread
  __shared__ __align__(16) __bf16 As[2][BM * LDK];
  __shared__ __align__(16) __bf16 Bs[2][BN * LDK];

  const int tid = threadIdx.x;
  const int wave = tid >> 5, lane = tid & 31;
  const int laneM = lane & 15, hic = lane >> 4;      // half-wave id
  const int m0 = blockIdx.y * BM, n0 = blockIdx.x * BN;
  const int wr = wave >> 1, wc = wave & 1;           // 4x2 wave grid

  v8f vz = {};
  v8f acc[2][4];
#pragma unroll
  for (int i = 0; i < 2; ++i)
#pragma unroll
    for (int j = 0; j < 4; ++j) acc[i][j] = vz;

  const int lrow = tid >> 1;            // 0..127
  const int lcol = (tid & 1) * 32;      // halves: 0 or 32

  // Stage one 128x64 A-tile + 128x64 W-tile into LDS buffer `buf`.
  auto stage = [&](int k0, int buf) {
    const __bf16* ag = &A[(m0 + lrow) * K + k0 + lcol];
    const __bf16* wg = &W[(n0 + lrow) * K + k0 + lcol];
    __bf16* as = &As[buf][lrow * LDK + lcol];
    __bf16* bs = &Bs[buf][lrow * LDK + lcol];
#if GEMM_ASYNC
#pragma unroll
    for (int c = 0; c < 4; ++c)
      __builtin_amdgcn_global_load_async_to_lds_b128(
          (v4i*)(ag + 8 * c), (AS3 v4i*)(as + 8 * c), 0, 0);
#pragma unroll
    for (int c = 0; c < 4; ++c)
      __builtin_amdgcn_global_load_async_to_lds_b128(
          (v4i*)(wg + 8 * c), (AS3 v4i*)(bs + 8 * c), 0, 0);
#else
    uint4 a0 = *(const uint4*)(ag + 0),  a1 = *(const uint4*)(ag + 8);
    uint4 a2 = *(const uint4*)(ag + 16), a3 = *(const uint4*)(ag + 24);
    uint4 b0 = *(const uint4*)(wg + 0),  b1 = *(const uint4*)(wg + 8);
    uint4 b2 = *(const uint4*)(wg + 16), b3 = *(const uint4*)(wg + 24);
    *(uint4*)(as + 0) = a0;  *(uint4*)(as + 8) = a1;
    *(uint4*)(as + 16) = a2; *(uint4*)(as + 24) = a3;
    *(uint4*)(bs + 0) = b0;  *(uint4*)(bs + 8) = b1;
    *(uint4*)(bs + 16) = b2; *(uint4*)(bs + 24) = b3;
#endif
  };

  const int steps = K / BK;             // 16
  stage(0, 0);

#pragma unroll 1
  for (int kt = 0; kt < steps; ++kt) {
    const int buf = kt & 1;
    if (kt + 1 < steps) {
      stage((kt + 1) * BK, buf ^ 1);    // overlap next tile with this compute
#if GEMM_ASYNC
      __builtin_amdgcn_s_wait_asynccnt(8);  // in-order: stage kt complete
#endif
    } else {
#if GEMM_ASYNC
      __builtin_amdgcn_s_wait_asynccnt(0);
#endif
    }
    __syncthreads();                    // stage kt visible to all waves

    const __bf16* asb = &As[buf][0];
    const __bf16* bsb = &Bs[buf][0];
#pragma unroll
    for (int s = 0; s < 2; ++s) {               // two 32-wide K steps
      const int ka = s * 32 + hic * 8;          // A-frag: k = ka..+7 and ka+16..+23
      const int kb = s * 32 + hic * 8;          // B-frag: k = kb..+7 and kb+16..+23
      v16bf af[2], bg[4];
#pragma unroll
      for (int mt = 0; mt < 2; ++mt) {
        const __bf16* r = &asb[(wr * 32 + mt * 16 + laneM) * LDK];
        af[mt] = ld_frag2(r + ka, r + ka + 16);
      }
#pragma unroll
      for (int nt = 0; nt < 4; ++nt) {
        const __bf16* r = &bsb[(wc * 64 + nt * 16 + laneM) * LDK];
        bg[nt] = ld_frag2(r + kb, r + kb + 16);
      }
#pragma unroll
      for (int mt = 0; mt < 2; ++mt)
#pragma unroll
        for (int nt = 0; nt < 4; ++nt)
          acc[mt][nt] = __builtin_amdgcn_wmma_f32_16x16x32_bf16(
              false, af[mt], false, bg[nt], (short)0, acc[mt][nt], false, false);
    }
    __syncthreads();                    // done reading buf before it is rewritten
  }

  // Epilogue: C-layout row = r + 8*hic, col = laneM
#pragma unroll
  for (int mt = 0; mt < 2; ++mt) {
#pragma unroll
    for (int nt = 0; nt < 4; ++nt) {
      const int n = n0 + wc * 64 + nt * 16 + laneM;
      const float bv = bias[n];
#pragma unroll
      for (int r = 0; r < 8; ++r) {
        const int m = m0 + wr * 32 + mt * 16 + r + hic * 8;
        const float v = acc[mt][nt][r] + bv;
        if constexpr (MODE == 0) {
          ((__bf16*)OutP)[m * N + n] = (__bf16)v;
        } else if constexpr (MODE == 1) {
          const int bloc = m >> 11, sloc = m & (S_ - 1);   // S_=2048
          const int nh = n >> 6, d = n & (HD_ - 1);
          ((__bf16*)OutP)[(((bloc * NH_ + nh) * HD_ + d) << 11) + sloc] = (__bf16)v;
        } else {
          ((float*)OutP)[m * N + n] = v;
        }
      }
    }
  }
}

// ---------------------------------------------------------------------------
// Causal flash attention. Grid = B*NH*(S/128), 256 threads (8 waves).
// Wave w owns query rows qblk*128 + w*16 .. +15, full HD=64 accumulator.
// Q,K: [B,S,H] bf16 (head h at col h*64). Vt: [B,NH,HD,S] bf16. Ctx: [B,S,H].
// No __syncthreads inside the causal loop (waves have different trip counts);
// the P C-layout -> A-layout fix-up goes through a wave-private LDS tile.
// Causal masking only on the diagonal block (wave-uniform branch).
// ---------------------------------------------------------------------------
__global__ __launch_bounds__(256) void attn_fwd(
    const __bf16* __restrict__ Q, const __bf16* __restrict__ Kmat,
    const __bf16* __restrict__ Vt, __bf16* __restrict__ Ctx) {
  __shared__ __align__(16) __bf16 Plds[8][16][40];  // per-wave P tile, padded rows

  const int tid = threadIdx.x, wave = tid >> 5, lane = tid & 31;
  const int laneM = lane & 15, hic = lane >> 4;
  const int bid = blockIdx.x;
  const int qblk = bid & 15;             // S/128 = 16
  const int h = (bid >> 4) & (NH_ - 1);
  const int b = bid >> 8;
  const int qw = qblk * 128 + wave * 16;

  const __bf16* qb = Q    + (size_t)b * S_ * H_ + h * HD_;
  const __bf16* kB = Kmat + (size_t)b * S_ * H_ + h * HD_;
  const __bf16* vb = Vt   + ((size_t)(b * NH_ + h)) * HD_ * S_;

  // Q fragments (reused across all key blocks): A-layout, contraction d=0..63
  const int qs = qw + laneM;
  v16bf qf[2];
#pragma unroll
  for (int s = 0; s < 2; ++s)
    qf[s] = ld_frag2(&qb[qs * H_ + s * 32 + hic * 8],
                     &qb[qs * H_ + s * 32 + 16 + hic * 8]);

  v8f vz = {};
  v8f acc[4];
#pragma unroll
  for (int t = 0; t < 4; ++t) acc[t] = vz;
  float mrow[8], lrow[8];
#pragma unroll
  for (int r = 0; r < 8; ++r) { mrow[r] = -1e30f; lrow[r] = 0.0f; }

  const float scale = 0.125f;  // 1/sqrt(64)

  for (int kb0 = 0; kb0 <= qw + 15; kb0 += 32) {   // wave-uniform bound
    // ---- scores: two 16-key tiles, QK^T over d (two K=32 WMMAs each) ----
    v8f sc[2];
#pragma unroll
    for (int j = 0; j < 2; ++j) {
      v8f c = vz;
      const int key = kb0 + j * 16 + laneM;        // B-frag col = key
#pragma unroll
      for (int s = 0; s < 2; ++s) {
        v16bf kf = ld_frag2(&kB[key * H_ + s * 32 + hic * 8],
                            &kB[key * H_ + s * 32 + hic * 8 + 16]);
        c = __builtin_amdgcn_wmma_f32_16x16x32_bf16(
            false, qf[s], false, kf, (short)0, c, false, false);
      }
      sc[j] = c;
    }

    // ---- scale (+ causal mask only on the diagonal block) ----
    float p0[8], p1[8];
    if (kb0 + 31 > qw) {                           // wave-uniform
      const int key0 = kb0 + laneM, key1 = kb0 + 16 + laneM;
#pragma unroll
      for (int r = 0; r < 8; ++r) {
        const int qrow = qw + r + hic * 8;
        float a = sc[0][r] * scale, bsc = sc[1][r] * scale;
        if (key0 > qrow) a   = -1e30f;
        if (key1 > qrow) bsc = -1e30f;
        p0[r] = a; p1[r] = bsc;
      }
    } else {
#pragma unroll
      for (int r = 0; r < 8; ++r) {
        p0[r] = sc[0][r] * scale;
        p1[r] = sc[1][r] * scale;
      }
    }

    // ---- online softmax (row reductions across the 16-lane group) ----
#pragma unroll
    for (int r = 0; r < 8; ++r) {
      float t = rmax16(fmaxf(p0[r], p1[r]));
      const float mn = fmaxf(mrow[r], t);
      const float alpha = __expf(mrow[r] - mn);
      mrow[r] = mn;
      p0[r] = __expf(p0[r] - mn);
      p1[r] = __expf(p1[r] - mn);
      lrow[r] = lrow[r] * alpha + rsum16(p0[r] + p1[r]);
#pragma unroll
      for (int t4 = 0; t4 < 4; ++t4) acc[t4][r] *= alpha;
      // C-layout -> LDS (row = r + 8*hic, cols laneM / 16+laneM)
      const int row = r + hic * 8;
      Plds[wave][row][laneM]      = (__bf16)p0[r];
      Plds[wave][row][16 + laneM] = (__bf16)p1[r];
    }

    // ---- P (A-layout from LDS) @ V (B-frags contiguous from Vt) ----
    v16bf pf = ld_frag2(&Plds[wave][laneM][hic * 8],
                        &Plds[wave][laneM][16 + hic * 8]);
#pragma unroll
    for (int t4 = 0; t4 < 4; ++t4) {
      const __bf16* vr = &vb[(t4 * 16 + laneM) * S_ + kb0 + hic * 8];
      v16bf vf = ld_frag2(vr, vr + 16);
      acc[t4] = __builtin_amdgcn_wmma_f32_16x16x32_bf16(
          false, pf, false, vf, (short)0, acc[t4], false, false);
    }
  }

  // ---- normalize + store ctx [B,S,H] bf16 ----
#pragma unroll
  for (int r = 0; r < 8; ++r) {
    const float rl = 1.0f / lrow[r];
    const int srow = qw + r + hic * 8;
#pragma unroll
    for (int t4 = 0; t4 < 4; ++t4)
      Ctx[(size_t)(b * S_ + srow) * H_ + h * HD_ + t4 * 16 + laneM] =
          (__bf16)(acc[t4][r] * rl);
  }
}

// ---------------------------------------------------------------------------
// Launch
// ---------------------------------------------------------------------------
extern "C" void kernel_launch(void* const* d_in, const int* in_sizes, int n_in,
                              void* d_out, int out_size, void* d_ws, size_t ws_size,
                              hipStream_t stream) {
  const float* X  = (const float*)d_in[0];
  const float* Wq = (const float*)d_in[1];
  const float* bq = (const float*)d_in[2];
  const float* Wk = (const float*)d_in[3];
  const float* bk = (const float*)d_in[4];
  const float* Wv = (const float*)d_in[5];
  const float* bv = (const float*)d_in[6];
  const float* Wo = (const float*)d_in[7];
  const float* bo = (const float*)d_in[8];

  char* ws = (char*)d_ws;
  const size_t XB = (size_t)M_ * H_ * 2;   // 8 MiB
  const size_t WB = (size_t)H_ * H_ * 2;   // 2 MiB
  __bf16* Xb   = (__bf16*)(ws);
  __bf16* Wqb  = (__bf16*)(ws + XB);
  __bf16* Wkb  = (__bf16*)(ws + XB + WB);
  __bf16* Wvb  = (__bf16*)(ws + XB + 2 * WB);
  __bf16* Wob  = (__bf16*)(ws + XB + 3 * WB);
  __bf16* Qb   = (__bf16*)(ws + XB + 4 * WB);
  __bf16* Kb   = (__bf16*)(ws + 2 * XB + 4 * WB);
  __bf16* Vtb  = (__bf16*)(ws + 3 * XB + 4 * WB);
  __bf16* Ctxb = (__bf16*)(ws + 4 * XB + 4 * WB);

  // f32 -> bf16
  const int nX4 = (M_ * H_) / 4, nW4 = (H_ * H_) / 4;
  cvt_f32_bf16_x4<<<(nX4 + 255) / 256, 256, 0, stream>>>(X, Xb, nX4);
  cvt_f32_bf16_x4<<<(nW4 + 255) / 256, 256, 0, stream>>>(Wq, Wqb, nW4);
  cvt_f32_bf16_x4<<<(nW4 + 255) / 256, 256, 0, stream>>>(Wk, Wkb, nW4);
  cvt_f32_bf16_x4<<<(nW4 + 255) / 256, 256, 0, stream>>>(Wv, Wvb, nW4);
  cvt_f32_bf16_x4<<<(nW4 + 255) / 256, 256, 0, stream>>>(Wo, Wob, nW4);

  // QKV projections
  dim3 gg(H_ / 128, M_ / 128);
  gemm_xwT<0><<<gg, 256, 0, stream>>>(Xb, Wqb, bq, Qb,  M_, H_, H_);
  gemm_xwT<0><<<gg, 256, 0, stream>>>(Xb, Wkb, bk, Kb,  M_, H_, H_);
  gemm_xwT<1><<<gg, 256, 0, stream>>>(Xb, Wvb, bv, Vtb, M_, H_, H_);

  // causal flash attention
  attn_fwd<<<B_ * NH_ * (S_ / 128), 256, 0, stream>>>(Qb, Kb, Vtb, Ctxb);

  // output projection -> f32 d_out
  gemm_xwT<2><<<gg, 256, 0, stream>>>(Ctxb, Wob, bo, (float*)d_out, M_, H_, H_);
}
